// Attention_85607288144195
// MI455X (gfx1250) — matrix operand order repaired and checked by reading, not compiled, not run on hardware
//
#include <hip/hip_runtime.h>
#include <hip/hip_bf16.h>
#include <math.h>

// ---------------------------------------------------------------------------
// Transformer-XL style relative attention for MI455X (gfx1250, wave32, WMMA).
// All GEMMs run through v_wmma_f32_16x16x32_bf16 (bf16 operands, f32 accum).
// Tiles are staged global->LDS with CDNA5 async copies
// (global_load_async_to_lds_b128 / s_wait_asynccnt) in a double-buffered
// pipeline: one barrier per K-step, next tile's copy overlaps the WMMAs.
// Shapes from the harness are multiples of the 128x128x32 tiling (B=8,
// Q=KV=2048, H=1024, R=4096), so no edge guards are needed.
// ---------------------------------------------------------------------------

#define USE_ASYNC_LDS 1   // set to 0 to fall back to load+ds_store staging

typedef __bf16 bf16_t;
typedef __attribute__((ext_vector_type(8)))  __bf16 bf16x8;
typedef __attribute__((ext_vector_type(16))) __bf16 bf16x16;
typedef __attribute__((ext_vector_type(8)))  float  f32x8;

#define BM 128
#define BN 128
#define BK 32
#define LDS_STRIDE 48   // 32 data bf16 + 16 pad; 96B row stride (16B aligned, bank-spread)

#if USE_ASYNC_LDS
// One 16B per-lane async copy: data never touches VGPRs, tracked by ASYNCcnt.
static __device__ __forceinline__ void async_b128(const void* g, void* l) {
  unsigned lds = (unsigned)(unsigned long long)(size_t)l;  // low 32b of generic ptr = LDS offset
  unsigned long long ga = (unsigned long long)(size_t)g;
  asm volatile("global_load_async_to_lds_b128 %0, %1, off"
               :: "v"(lds), "v"(ga) : "memory");
}
static __device__ __forceinline__ void wait_async0() {
  asm volatile("s_wait_asynccnt 0x0" ::: "memory");
}
#endif

// -------------------------------- converts --------------------------------

__global__ void f32_to_bf16_kernel(const float* __restrict__ src,
                                   bf16_t* __restrict__ dst, size_t n) {
  size_t i = (size_t)blockIdx.x * blockDim.x + threadIdx.x;
  if (i < n) dst[i] = (bf16_t)src[i];
}

// key -> bf16 copy (B,KV,H) and bf16 transposed copy (B,H,KV)
__global__ void key_convert_kernel(const float* __restrict__ key,
                                   bf16_t* __restrict__ kb,
                                   bf16_t* __restrict__ ktb,
                                   int KV, int H, size_t n) {
  size_t i = (size_t)blockIdx.x * blockDim.x + threadIdx.x;
  if (i >= n) return;
  int h = (int)(i % H);
  size_t t = i / H;
  int k = (int)(t % KV);
  size_t b = t / KV;
  float v = key[i];
  kb[i] = (bf16_t)v;
  ktb[(b * H + h) * (size_t)KV + k] = (bf16_t)v;
}

// kr[r][h]: sin/cos((KV-r) * 10000^(-2*hh/H)),  hh = h mod H/2
__global__ void kr_kernel(bf16_t* __restrict__ kr, int H, int KV,
                          float log_base_over_H, size_t n) {
  size_t i = (size_t)blockIdx.x * blockDim.x + threadIdx.x;
  if (i >= n) return;
  int h = (int)(i % H);
  int r = (int)(i / H);
  int Hh = H >> 1;
  int hh = (h < Hh) ? h : h - Hh;
  float invf = expf(-2.0f * (float)hh * log_base_over_H);
  float ang = (float)(KV - r) * invf;
  kr[i] = (bf16_t)((h < Hh) ? sinf(ang) : cosf(ang));
}

// ---------------------- generic batched NT WMMA GEMM -----------------------
// C[M,N] = A[M,K] * B[N,K]^T (+ bias[N]).  Batched via blockIdx.z + strides.
// BAND: skip output tiles outside r in [bandQ-m0-127, bandQ-m0+bandKV-1]
// (only those entries of the position-score matrix survive rel_shift).

template <int BF16OUT, int HASBIAS, int BAND>
__global__ void __launch_bounds__(256)
gemm_nt_kernel(const bf16_t* __restrict__ Ag, const bf16_t* __restrict__ Bg,
               const float* __restrict__ bias, void* __restrict__ Cg,
               int M, int N, int K,
               long long sA, long long sB, long long sC,
               int bandQ, int bandKV) {
  const int m0 = blockIdx.y * BM;
  const int n0 = blockIdx.x * BN;

  if (BAND) {
    int lo = bandQ - (m0 + BM - 1);
    int hi = bandQ - m0 + bandKV - 1;
    if ((n0 + BN - 1) < lo || n0 > hi) return;  // uniform across the block
  }

  const size_t bz = blockIdx.z;
  const bf16_t* A  = Ag + bz * (size_t)sA;
  const bf16_t* Bm = Bg + bz * (size_t)sB;

  __shared__ bf16_t As[2][BM * LDS_STRIDE];   // double-buffered tiles
  __shared__ bf16_t Bs[2][BN * LDS_STRIDE];

  const int tid  = threadIdx.x;
  const int lane = tid & 31;
  const int wid  = tid >> 5;           // 8 waves
  const int wrow = (wid >> 1) * 32;    // wave row base: 0/32/64/96
  const int wcol = (wid & 1) * 64;     // wave col base: 0/64
  const int half = lane >> 4;          // 0/1 (lane group)
  const int l16  = lane & 15;

  const int ldrow  = tid >> 1;         // 0..127 (staging row)
  const int ldhalf = (tid & 1) * 16;   // 0/16   (staging K half)
  const int ldoff  = ldrow * LDS_STRIDE + ldhalf;

  f32x8 acc[2][4];
  const f32x8 fz = {0.f, 0.f, 0.f, 0.f, 0.f, 0.f, 0.f, 0.f};
#pragma unroll
  for (int i = 0; i < 2; ++i)
#pragma unroll
    for (int j = 0; j < 4; ++j) acc[i][j] = fz;

#if USE_ASYNC_LDS
  // 4 fire-and-forget 16B copies per thread per K-step (A tile + B tile)
  auto stage_async = [&](int kb, int buf) {
    const bf16_t* sa = A  + (size_t)(m0 + ldrow) * K + kb + ldhalf;
    const bf16_t* sb = Bm + (size_t)(n0 + ldrow) * K + kb + ldhalf;
    bf16_t* da = &As[buf][ldoff];
    bf16_t* db = &Bs[buf][ldoff];
    async_b128(sa,     da);
    async_b128(sa + 8, da + 8);
    async_b128(sb,     db);
    async_b128(sb + 8, db + 8);
  };
  stage_async(0, 0);
#else
  bf16x8 ra0, ra1, rb0, rb1;
  auto gload = [&](int kb) {
    const bf16_t* sa = A  + (size_t)(m0 + ldrow) * K + kb + ldhalf;
    const bf16_t* sb = Bm + (size_t)(n0 + ldrow) * K + kb + ldhalf;
    ra0 = *reinterpret_cast<const bf16x8*>(sa);
    ra1 = *reinterpret_cast<const bf16x8*>(sa + 8);
    rb0 = *reinterpret_cast<const bf16x8*>(sb);
    rb1 = *reinterpret_cast<const bf16x8*>(sb + 8);
  };
  auto lstore = [&](int buf) {
    bf16_t* da = &As[buf][ldoff];
    bf16_t* db = &Bs[buf][ldoff];
    *reinterpret_cast<bf16x8*>(da)     = ra0;
    *reinterpret_cast<bf16x8*>(da + 8) = ra1;
    *reinterpret_cast<bf16x8*>(db)     = rb0;
    *reinterpret_cast<bf16x8*>(db + 8) = rb1;
  };
  gload(0);
#endif

  int cur = 0;
  for (int kk = 0; kk < K; kk += BK) {
#if USE_ASYNC_LDS
    wait_async0();        // this wave's copies into buf[cur] have landed
    __syncthreads();      // every wave waited + finished last iter's reads
    if (kk + BK < K) stage_async(kk + BK, cur ^ 1);  // overlaps the WMMAs
#else
    lstore(cur);                       // stores of regs loaded last iter
    if (kk + BK < K) gload(kk + BK);   // next tile's loads overlap WMMAs
    __syncthreads();                   // buf[cur] visible to all waves
#endif

    // A fragments: lane<16 holds row M=l16, K {0..7,16..23}; lanes 16..31
    // hold K {8..15,24..31} (ISA 16-bit A layout).
    bf16x16 af[2], bfr[4];
#pragma unroll
    for (int i = 0; i < 2; ++i) {
      const bf16_t* p = &As[cur][(wrow + i * 16 + l16) * LDS_STRIDE + half * 8];
      bf16x8 lo = *reinterpret_cast<const bf16x8*>(p);
      bf16x8 hi = *reinterpret_cast<const bf16x8*>(p + 16);
#pragma unroll
      for (int e = 0; e < 8; ++e) { af[i][e] = lo[e]; af[i][8 + e] = hi[e]; }
    }
    // B fragments: lane = column n, lane<16 K {0..7,16..23}, lanes 16..31 K {8..15,24..31}
#pragma unroll
    for (int j = 0; j < 4; ++j) {
      const bf16_t* p = &Bs[cur][(wcol + j * 16 + l16) * LDS_STRIDE + half * 8];
      bf16x8 lo = *reinterpret_cast<const bf16x8*>(p);
      bf16x8 hi = *reinterpret_cast<const bf16x8*>(p + 16);
#pragma unroll
      for (int e = 0; e < 8; ++e) { bfr[j][e] = lo[e]; bfr[j][8 + e] = hi[e]; }
    }

#pragma unroll
    for (int i = 0; i < 2; ++i)
#pragma unroll
      for (int j = 0; j < 4; ++j)
        acc[i][j] = __builtin_amdgcn_wmma_f32_16x16x32_bf16(
            false, af[i], false, bfr[j], (short)0, acc[i][j], false, false);

    cur ^= 1;
  }

  // epilogue: C/D layout VGPR v -> M = v + 8*half, N = l16
#pragma unroll
  for (int i = 0; i < 2; ++i) {
#pragma unroll
    for (int j = 0; j < 4; ++j) {
#pragma unroll
      for (int v = 0; v < 8; ++v) {
        int row = m0 + wrow + i * 16 + half * 8 + v;
        int col = n0 + wcol + j * 16 + l16;
        float val = acc[i][j][v];
        if (HASBIAS) val += bias[col];
        if (BF16OUT) {
          bf16_t* C = reinterpret_cast<bf16_t*>(Cg) + bz * (size_t)sC;
          C[(size_t)row * N + col] = (bf16_t)val;
        } else {
          float* C = reinterpret_cast<float*>(Cg) + bz * (size_t)sC;
          C[(size_t)row * N + col] = val;
        }
      }
    }
  }
}

// ------------------- combine + rel_shift + mask + softmax ------------------
// scores(b,q,k) = inv_scale * (content(b,q,k) + P(b,q, k+Q-q)); masked -> -inf
// weights written back over content in d_out, bf16 copy to wb for final GEMM.

__global__ void combine_softmax_kernel(const float* __restrict__ P,
                                       const unsigned char* __restrict__ mask,
                                       float* __restrict__ outW,
                                       bf16_t* __restrict__ wb,
                                       int Q, int KV, int R, float inv_scale) {
  const int q = blockIdx.x;
  const size_t rowoff = (size_t)blockIdx.y * Q + q;
  const float* crow = outW + rowoff * KV;
  const float* prow = P + rowoff * R + (Q - q);   // rel_shift index map
  const unsigned char* mrow = mask + rowoff * KV;
  float*  wrow  = outW + rowoff * KV;
  bf16_t* wbrow = wb + rowoff * KV;

  __shared__ float red[256];
  const int t = threadIdx.x;

  // pass 1: row max
  float mx = -INFINITY;
  for (int k = t; k < KV; k += 256) {
    float s = mrow[k] ? -INFINITY : (crow[k] + prow[k]) * inv_scale;
    mx = fmaxf(mx, s);
  }
  red[t] = mx; __syncthreads();
  for (int o = 128; o > 0; o >>= 1) {
    if (t < o) red[t] = fmaxf(red[t], red[t + o]);
    __syncthreads();
  }
  mx = red[0]; __syncthreads();

  // pass 2: exp + sum (unnormalized exp overwrites content in place)
  float sum = 0.f;
  for (int k = t; k < KV; k += 256) {
    float s = mrow[k] ? -INFINITY : (crow[k] + prow[k]) * inv_scale;
    float e = expf(s - mx);
    wrow[k] = e;
    sum += e;
  }
  red[t] = sum; __syncthreads();
  for (int o = 128; o > 0; o >>= 1) {
    if (t < o) red[t] += red[t + o];
    __syncthreads();
  }
  const float inv = 1.f / red[0];

  // pass 3: normalize + bf16 copy
  for (int k = t; k < KV; k += 256) {
    float w = wrow[k] * inv;
    wrow[k] = w;
    wbrow[k] = (bf16_t)w;
  }
}

// --------------------------------- launch ----------------------------------

extern "C" void kernel_launch(void* const* d_in, const int* in_sizes, int n_in,
                              void* d_out, int out_size, void* d_ws, size_t ws_size,
                              hipStream_t stream) {
  const float* query = (const float*)d_in[0];
  const float* key   = (const float*)d_in[1];
  const unsigned char* mask = (const unsigned char*)d_in[2];  // jax bool = 1B
  const float* Wc = (const float*)d_in[3];
  const float* Wp = (const float*)d_in[4];
  const float* bp = (const float*)d_in[5];

  const int       H  = in_sizes[5];
  const long long BQ = (long long)in_sizes[0] / H;
  const int       KV = (int)((long long)in_sizes[2] / BQ);
  const int       B  = (int)((long long)in_sizes[1] / ((long long)KV * H));
  const int       Q  = (int)(BQ / B);
  const int       R  = KV + Q;
  const float inv_scale = 1.0f / sqrtf((float)H);

  // workspace carve-up (256B aligned)
  char* wp_ = (char*)d_ws;
  auto carve = [&](size_t bytes) -> char* {
    char* r = wp_;
    wp_ += (bytes + 255) & ~(size_t)255;
    return r;
  };
  const size_t nQ = (size_t)B * Q * H;    // query elems
  const size_t nK = (size_t)B * KV * H;   // key elems
  const size_t nW = (size_t)B * Q * KV;   // score/weight elems
  const size_t nR = (size_t)R * H;        // kr elems

  bf16_t* qb  = (bf16_t*)carve(nQ * 2);          // query bf16
  bf16_t* kb  = (bf16_t*)carve(nK * 2);          // key bf16 (B,KV,H)
  bf16_t* ktb = (bf16_t*)carve(nK * 2);          // key^T bf16 (B,H,KV)
  bf16_t* wcb = (bf16_t*)carve((size_t)H * H * 2);
  bf16_t* wpb = (bf16_t*)carve((size_t)H * H * 2);
  bf16_t* krb = (bf16_t*)carve(nR * 2);          // kr bf16 (R,H)
  bf16_t* qhb = (bf16_t*)carve(nQ * 2);          // content proj bf16
  bf16_t* qrb = (bf16_t*)carve(nQ * 2);          // position proj bf16
  float*  P   = (float*) carve((size_t)B * Q * R * 4);  // raw position scores
  bf16_t* wb  = (bf16_t*)carve(nW * 2);          // softmax weights bf16
  (void)ws_size;

  float* out_f   = (float*)d_out;                // (B,Q,H)
  float* weights = out_f + nQ;                   // (B,Q,KV)

  const int TPB = 256;
  auto blocks = [](size_t n) { return (unsigned)((n + 255) / 256); };

  // 1) precision converts + transposed key + sinusoid table
  f32_to_bf16_kernel<<<blocks(nQ), TPB, 0, stream>>>(query, qb, nQ);
  key_convert_kernel<<<blocks(nK), TPB, 0, stream>>>(key, kb, ktb, KV, H, nK);
  f32_to_bf16_kernel<<<blocks((size_t)H * H), TPB, 0, stream>>>(Wc, wcb, (size_t)H * H);
  f32_to_bf16_kernel<<<blocks((size_t)H * H), TPB, 0, stream>>>(Wp, wpb, (size_t)H * H);
  kr_kernel<<<blocks(nR), TPB, 0, stream>>>(krb, H, KV, logf(10000.0f) / H, nR);

  // 2) projections: (B*Q,H) x (H,H)^T  -> bf16
  {
    dim3 g(H / BN, (unsigned)(BQ / BM), 1);
    gemm_nt_kernel<1, 0, 0><<<g, TPB, 0, stream>>>(
        qb, wcb, nullptr, qhb, (int)BQ, H, H, 0, 0, 0, 0, 0);
    gemm_nt_kernel<1, 1, 0><<<g, TPB, 0, stream>>>(
        qb, wpb, bp, qrb, (int)BQ, H, H, 0, 0, 0, 0, 0);
  }

  // 3) content scores -> weights slot of d_out (raw, f32)
  {
    dim3 g(KV / BN, Q / BM, B);
    gemm_nt_kernel<0, 0, 0><<<g, TPB, 0, stream>>>(
        qhb, kb, nullptr, weights, Q, KV, H,
        (long long)Q * H, (long long)KV * H, (long long)Q * KV, 0, 0);
  }

  // 4) position scores P = qr x kr^T, band-limited to entries rel_shift reads
  {
    dim3 g(R / BN, Q / BM, B);
    gemm_nt_kernel<0, 0, 1><<<g, TPB, 0, stream>>>(
        qrb, krb, nullptr, P, Q, R, H,
        (long long)Q * H, 0, (long long)Q * R, Q, KV);
  }

  // 5) combine + rel_shift + mask + softmax (in place over weights) + bf16 copy
  {
    dim3 g(Q, B, 1);
    combine_softmax_kernel<<<g, TPB, 0, stream>>>(P, mask, weights, wb,
                                                  Q, KV, R, inv_scale);
  }

  // 6) out = weights x key : (Q,KV) x (H,KV)^T per batch
  {
    dim3 g(H / BN, Q / BM, B);
    gemm_nt_kernel<0, 0, 0><<<g, TPB, 0, stream>>>(
        wb, ktb, nullptr, out_f, Q, H, KV,
        (long long)Q * KV, (long long)H * KV, (long long)Q * H, 0, 0);
  }
}
